// ClusterKT_7481833030183
// MI455X (gfx1250) — matrix operand order repaired and checked by reading, not compiled, not run on hardware
//
#include <hip/hip_runtime.h>

// ---------------- problem constants (from reference) ----------------
#define B_    4
#define L_    1024
#define D_    512
#define H_    8
#define DK_   64
#define DFF_  2048
#define EPS_  1e-5f
#define NEG_  (-1e32f)
#define SCALE_ 0.125f   // 1/sqrt(64)

typedef __attribute__((ext_vector_type(16))) __bf16 v16bf;
typedef __attribute__((ext_vector_type(8)))  float  v8f;
typedef __attribute__((ext_vector_type(4)))  unsigned int v4u;
typedef __attribute__((ext_vector_type(8)))  unsigned int v8u;

// ---------------- wave32 reduction helpers ----------------
__device__ __forceinline__ float wave_max_f(float v) {
#pragma unroll
  for (int d = 16; d > 0; d >>= 1) v = fmaxf(v, __shfl_xor(v, d, 32));
  return v;
}
__device__ __forceinline__ float wave_sum_f(float v) {
#pragma unroll
  for (int d = 16; d > 0; d >>= 1) v += __shfl_xor(v, d, 32);
  return v;
}

// ---------------- vectorized WMMA fragment builders (CDNA5 16x16x32 bf16) ----------------
// A (16x32 MxK): lane (half,m) loads K = {kk+16h+0..7} in VGPR0-3 and {kk+16h+8..15}
// in VGPR4-7 -> two contiguous 16B loads (global_load_b128 / ds_load_b128).
__device__ __forceinline__ v16bf frag_a_vec(const __bf16* rowp, int kk, int half) {
  union { v4u u[2]; v16bf f; } c;
  c.u[0] = *(const v4u*)(rowp + kk + 16 * half);
  c.u[1] = *(const v4u*)(rowp + kk + 16 * half + 8);
  return c.f;
}

// B (32x16 KxN) with K-contiguous source per column (B^T layout): lane (half,n) needs
// K = {base+0..15} contiguous -> one 32B load (2x b128).
__device__ __forceinline__ v16bf frag_b_vec(const __bf16* p) {
  return *(const v16bf*)p;
}

__device__ __forceinline__ v8f wmma_bf16(v16bf a, v16bf b, v8f c) {
  return __builtin_amdgcn_wmma_f32_16x16x32_bf16(false, a, false, b, (short)0, c, false, false);
}

// ---------------- f32 -> bf16 conversion ----------------
__global__ void cvt_f32_bf16(const float* __restrict__ in, __bf16* __restrict__ out, int n) {
  int i = blockIdx.x * blockDim.x + threadIdx.x;
  if (i < n) out[i] = (__bf16)in[i];
}

// ---------------- f32 [K][N] -> bf16 transposed [N][K] (for GEMM B operand) ----------------
__global__ void cvt_transpose_bf16(const float* __restrict__ in, __bf16* __restrict__ out,
                                   int K, int N) {
  __shared__ float tile[32][33];
  const int kb = blockIdx.y * 32;
  const int nb = blockIdx.x * 32;
  const int tx = threadIdx.x;    // 32
  const int ty = threadIdx.y;    // 8
  for (int r = ty; r < 32; r += 8)
    tile[r][tx] = in[(size_t)(kb + r) * N + nb + tx];
  __syncthreads();
  for (int r = ty; r < 32; r += 8)
    out[(size_t)(nb + r) * K + kb + tx] = (__bf16)tile[tx][r];
}

// ---------------- generic bf16 WMMA GEMM (B pre-transposed to [N][K]) ----------------
// C[M,N] = A[M,K] @ B[K,N] + bias
// flags: 1 = ReLU; 2 = bf16 out in (B,H,L,DK); 4 = bf16 out in (B,H,DK,L) (V^T)
// 256 threads = 8 waves; each wave owns TWO 16x16 tiles sharing one A fragment.
__global__ void gemm_bf16_wmma(const __bf16* __restrict__ A, const __bf16* __restrict__ Bt,
                               const float* __restrict__ bias,
                               float* __restrict__ outF, __bf16* __restrict__ outB,
                               int M, int N, int K, int flags) {
  const int lane = threadIdx.x & 31;
  const int wave = threadIdx.x >> 5;
  const int half = lane >> 4;
  const int rm   = lane & 15;
  const int m0 = blockIdx.y * 16;
  const int n0 = blockIdx.x * 256 + wave * 32;
  const int n1 = n0 + 16;

  const __bf16* arow  = A  + (size_t)(m0 + rm) * K;
  const __bf16* brow0 = Bt + (size_t)(n0 + rm) * K;
  const __bf16* brow1 = Bt + (size_t)(n1 + rm) * K;

  v8f acc0 = {}, acc1 = {};
  for (int kk = 0; kk < K; kk += 32) {
    __builtin_prefetch(arow + kk + 256, 0, 3);   // global_prefetch_b8
    const v16bf af = frag_a_vec(arow, kk, half);
    const v16bf b0 = frag_b_vec(brow0 + kk + 16 * half);
    const v16bf b1 = frag_b_vec(brow1 + kk + 16 * half);
    acc0 = wmma_bf16(af, b0, acc0);
    acc1 = wmma_bf16(af, b1, acc1);
  }

#pragma unroll
  for (int t = 0; t < 2; ++t) {
    const v8f acc = t ? acc1 : acc0;
    const int ncol = (t ? n1 : n0) + rm;
    const float bn = bias ? bias[ncol] : 0.0f;
    const int h = ncol / DK_, dk = ncol % DK_;
#pragma unroll
    for (int r = 0; r < 8; ++r) {
      const int mrow = m0 + r + 8 * half;   // C layout: VGPR r -> M = r + 8*half
      float v = acc[r] + bn;
      if (flags & 1) v = fmaxf(v, 0.0f);
      if (outF) outF[(size_t)mrow * N + ncol] = v;
      if (outB) {
        const int b = mrow / L_, i = mrow % L_;
        size_t idx;
        if (flags & 2)      idx = (((size_t)b * H_ + h) * L_ + i) * DK_ + dk;      // (B,H,L,DK)
        else if (flags & 4) idx = (((size_t)b * H_ + h) * DK_ + dk) * L_ + i;      // (B,H,DK,L)
        else                idx = (size_t)mrow * N + ncol;
        outB[idx] = (__bf16)v;
      }
    }
  }
}

// ---------------- fused forgetting-attention ----------------
// grid = (L/16, B*H), block = 256 (8 waves).
// Dynamic LDS: S[16][L] f32 (64KB) + P[16][L] bf16 (32KB) + Kpanel[L][DK] bf16 (128KB).
// The K panel is staged by the Tensor Data Mover (tensor_load_to_lds, TENSORcnt).
#define ATTN_S_BYTES   ((size_t)16 * L_ * sizeof(float))
#define ATTN_P_BYTES   ((size_t)16 * L_ * sizeof(__bf16))
#define ATTN_KLDS_OFF  (ATTN_S_BYTES + ATTN_P_BYTES)          // 98304
#define ATTN_K_BYTES   ((size_t)L_ * DK_ * sizeof(__bf16))    // 131072
__global__ void fused_attn(const __bf16* __restrict__ Qh, const __bf16* __restrict__ Kh,
                           const __bf16* __restrict__ Vt,
                           const float* __restrict__ satate, const float* __restrict__ decay,
                           __bf16* __restrict__ ctx) {
  extern __shared__ char smem[];
  float*  S    = (float*)smem;                          // scores tile
  __bf16* Pa   = (__bf16*)(smem + ATTN_S_BYTES);        // attn tile (bf16)
  __bf16* Klds = (__bf16*)(smem + ATTN_KLDS_OFF);       // K panel staged by TDM

  const int bh = blockIdx.y;
  const int b  = bh / H_;
  const int h  = bh % H_;
  const int i0 = blockIdx.x * 16;
  const int lane = threadIdx.x & 31;
  const int wave = threadIdx.x >> 5;
  const int half = lane >> 4;
  const int rm   = lane & 15;

  const __bf16* Qb = Qh + (size_t)bh * L_ * DK_;
  const __bf16* Kb = Kh + (size_t)bh * L_ * DK_;
  const __bf16* Vb = Vt + (size_t)bh * DK_ * L_;

  // ---- TDM: DMA the whole per-head K panel (L x DK bf16, contiguous) into LDS ----
  if (wave == 0) {
    const unsigned long long ga = (unsigned long long)(const void*)Kb;
    v4u g0;
    g0[0] = 1u;                                            // count=1, user descriptor
    g0[1] = (unsigned)ATTN_KLDS_OFF;                       // lds_addr (bytes)
    g0[2] = (unsigned)ga;                                  // global_addr[31:0]
    g0[3] = (unsigned)((ga >> 32) & 0x01FFFFFFu) | (2u << 30);  // addr[56:32] | type=2
    v8u g1;
    g1[0] = 1u << 16;                                      // data_size=1 (2 bytes)
    g1[1] = (unsigned)DK_ << 16;                           // tensor_dim0[15:0] @ bits63:48
    g1[2] = ((unsigned)DK_ >> 16) | ((unsigned)L_ << 16);  // dim0 hi | tensor_dim1 lo
    g1[3] = ((unsigned)L_ >> 16) | ((unsigned)DK_ << 16);  // dim1 hi | tile_dim0=64
    g1[4] = (unsigned)L_;                                  // tile_dim1=1024 (tile_dim2=0)
    g1[5] = (unsigned)DK_;                                 // tensor_dim0_stride lo
    g1[6] = 0u;                                            // stride0 hi | dim1_stride lo
    g1[7] = 0u;
    asm volatile("tensor_load_to_lds %0, %1" :: "s"(g0), "s"(g1) : "memory");
    __builtin_amdgcn_s_wait_tensorcnt((short)0);
  }

  // ---- Phase 1 A-fragments (Q rows) load from global while the TDM runs ----
  const __bf16* qrow = Qb + (size_t)(i0 + rm) * DK_;
  const v16bf a0 = frag_a_vec(qrow, 0, half);
  const v16bf a1 = frag_a_vec(qrow, 32, half);
  __syncthreads();   // K panel visible to all waves

  // ---- Phase 1: S[16][L] = (Q_tile @ K^T) * SCALE via WMMA (K from LDS) ----
  for (int jt = wave * 8; jt < wave * 8 + 8; ++jt) {  // 64 j-tiles over 8 waves
    const int j0 = jt * 16;
    const __bf16* krow = Klds + (size_t)(j0 + rm) * DK_;   // ds_load_b128
    v8f c = {};
    c = wmma_bf16(a0, frag_b_vec(krow + 16 * half), c);
    c = wmma_bf16(a1, frag_b_vec(krow + 32 + 16 * half), c);
#pragma unroll
    for (int r = 0; r < 8; ++r)
      S[(size_t)(r + 8 * half) * L_ + j0 + rm] = c[r] * SCALE_;
  }
  __syncthreads();

  // ---- Phase 2: per-row masked softmax -> cumsum -> decay effect -> softmax ----
  for (int rr = 0; rr < 2; ++rr) {
    const int r  = wave + rr * 8;     // each wave owns 2 of 16 rows
    const int ig = i0 + r;
    float* Srow = S + (size_t)r * L_;
    const float* ssrow = satate + ((size_t)b * L_ + ig) * L_;
    const float* dcrow = decay  + ((size_t)b * L_ + ig) * L_;

    // softmax 1 over strictly-causal mask (j < ig)
    float m1 = NEG_;
    for (int j = lane; j < L_; j += 32)
      if (j < ig) m1 = fmaxf(m1, Srow[j]);
    m1 = wave_max_f(m1);

    float ta = 0.0f, tm = 0.0f;
    for (int j = lane; j < L_; j += 32) {
      const float sv = (j < ig) ? Srow[j] : NEG_;
      const float e  = __expf(sv - m1);
      ta += e;
      if (j < ig) tm += e;
    }
    ta = wave_sum_f(ta);
    tm = wave_sum_f(tm);
    const float invta = 1.0f / ta;
    const float dtot  = tm * invta;        // disttotal = sum(p)

    // inclusive cumsum of p + decay effect, rewrite Srow with scores*effect
    float run = 0.0f, m2 = NEG_;
    for (int jb = 0; jb < L_; jb += 32) {
      const int j = jb + lane;
      const float sv = Srow[j];
      float sc = (j < ig) ? __expf(sv - m1) * invta : 0.0f;
#pragma unroll
      for (int d = 1; d < 32; d <<= 1) {   // wave inclusive scan
        const float t = __shfl_up(sc, d, 32);
        if (lane >= d) sc += t;
      }
      const float csum = run + sc;
      run += __shfl(sc, 31, 32);
      const float dist  = sqrtf(fmaxf((dtot - csum) * ssrow[j], 0.0f));
      const float gamma = -log1pf(__expf(dcrow[j]));   // -softplus(decay)
      float eff = __expf(dist * gamma);
      eff = fminf(fmaxf(eff, 1e-5f), 1e5f);
      const float s2 = (j < ig) ? sv * eff : NEG_;
      Srow[j] = s2;
      m2 = fmaxf(m2, s2);
    }
    m2 = wave_max_f(m2);

    float s2sum = 0.0f;
    for (int j = lane; j < L_; j += 32) s2sum += __expf(Srow[j] - m2);
    s2sum = wave_sum_f(s2sum);
    const float inv2 = 1.0f / s2sum;

    __bf16* Prow = Pa + (size_t)r * L_;
    for (int j = lane; j < L_; j += 32) {
      const float av = (ig == 0) ? 0.0f : __expf(Srow[j] - m2) * inv2; // zero_pad row 0
      Prow[j] = (__bf16)av;
    }
  }
  __syncthreads();

  // ---- Phase 3: ctx[16][64] = attn[16][L] @ V[L][64]; waves 0..3 own 16 cols each ----
  if (wave < 4) {
    const int n0 = wave * 16;
    const __bf16* prow = Pa + (size_t)rm * L_;          // LDS, contiguous K -> ds_load_b128
    const __bf16* vrow = Vb + (size_t)(n0 + rm) * L_;   // V^T row, contiguous K
    v8f c = {};
    for (int kk = 0; kk < L_; kk += 32) {
      const v16bf af = frag_a_vec(prow, kk, half);
      const v16bf bf = frag_b_vec(vrow + kk + 16 * half);
      c = wmma_bf16(af, bf, c);
    }
#pragma unroll
    for (int r = 0; r < 8; ++r) {
      const int ig = i0 + r + 8 * half;
      ctx[((size_t)b * L_ + ig) * D_ + h * DK_ + n0 + rm] = (__bf16)c[r];
    }
  }
}

// ---------------- residual + LayerNorm (one D=512 row per 256-thread block) ----------------
__global__ void add_layernorm(const float* __restrict__ X, const float* __restrict__ Y,
                              const float* __restrict__ g, const float* __restrict__ beta,
                              float* __restrict__ outF, __bf16* __restrict__ outB) {
  const int row = blockIdx.x;
  const int t = threadIdx.x;            // 256 threads, 2 elems each
  const float* xr = X + (size_t)row * D_;
  const float* yr = Y + (size_t)row * D_;
  const float x0 = xr[t] + yr[t];
  const float x1 = xr[t + 256] + yr[t + 256];

  __shared__ float red[8];
  float s = wave_sum_f(x0 + x1);
  if ((t & 31) == 0) red[t >> 5] = s;
  __syncthreads();
  float tot = 0.0f;
#pragma unroll
  for (int i = 0; i < 8; ++i) tot += red[i];
  const float mean = tot * (1.0f / D_);
  __syncthreads();

  const float v0 = x0 - mean, v1 = x1 - mean;
  float sv = wave_sum_f(v0 * v0 + v1 * v1);
  if ((t & 31) == 0) red[t >> 5] = sv;
  __syncthreads();
  float vtot = 0.0f;
#pragma unroll
  for (int i = 0; i < 8; ++i) vtot += red[i];
  const float inv = rsqrtf(vtot * (1.0f / D_) + EPS_);

  const float o0 = g[t] * v0 * inv + beta[t];
  const float o1 = g[t + 256] * v1 * inv + beta[t + 256];
  if (outF) { outF[(size_t)row * D_ + t] = o0; outF[(size_t)row * D_ + t + 256] = o1; }
  if (outB) { outB[(size_t)row * D_ + t] = (__bf16)o0; outB[(size_t)row * D_ + t + 256] = (__bf16)o1; }
}

// ---------------- host orchestration ----------------
static inline void* carve(char*& p, size_t bytes) {
  void* r = (void*)p;
  p += (bytes + 255) & ~(size_t)255;
  return r;
}

extern "C" void kernel_launch(void* const* d_in, const int* in_sizes, int n_in,
                              void* d_out, int out_size, void* d_ws, size_t ws_size,
                              hipStream_t stream) {
  (void)in_sizes; (void)n_in; (void)out_size; (void)ws_size;

  const float* query  = (const float*)d_in[0];
  const float* key_in = (const float*)d_in[1];
  const float* values = (const float*)d_in[2];
  const float* satate = (const float*)d_in[3];
  const float* decay  = (const float*)d_in[4];
  const float* Wq = (const float*)d_in[5];   const float* bq = (const float*)d_in[6];
  const float* Wk = (const float*)d_in[7];   const float* bk = (const float*)d_in[8];
  const float* Wv = (const float*)d_in[9];   const float* bv = (const float*)d_in[10];
  const float* Wo = (const float*)d_in[11];  const float* bo = (const float*)d_in[12];
  const float* ln1_g = (const float*)d_in[13]; const float* ln1_b = (const float*)d_in[14];
  const float* W1 = (const float*)d_in[15];  const float* b1 = (const float*)d_in[16];
  const float* W2 = (const float*)d_in[17];  const float* b2 = (const float*)d_in[18];
  const float* ln2_g = (const float*)d_in[19]; const float* ln2_b = (const float*)d_in[20];

  const size_t nX  = (size_t)B_ * L_ * D_;    // 2M
  const size_t nW  = (size_t)D_ * D_;         // 256K
  const size_t nW1 = (size_t)D_ * DFF_;       // 1M
  const size_t nFF = (size_t)B_ * L_ * DFF_;  // 8M

  char* wp = (char*)d_ws;
  __bf16* Xq_b = (__bf16*)carve(wp, nX * 2);
  __bf16* Xk_b = (__bf16*)carve(wp, nX * 2);
  __bf16* Xv_b = (__bf16*)carve(wp, nX * 2);
  __bf16* Wq_t = (__bf16*)carve(wp, nW * 2);   // transposed [N][K]
  __bf16* Wk_t = (__bf16*)carve(wp, nW * 2);
  __bf16* Wv_t = (__bf16*)carve(wp, nW * 2);
  __bf16* Wo_t = (__bf16*)carve(wp, nW * 2);
  __bf16* W1_t = (__bf16*)carve(wp, nW1 * 2);
  __bf16* W2_t = (__bf16*)carve(wp, nW1 * 2);
  __bf16* q_b  = (__bf16*)carve(wp, nX * 2);   // (B,H,L,DK)
  __bf16* k_b  = (__bf16*)carve(wp, nX * 2);   // (B,H,L,DK)
  __bf16* v_t  = (__bf16*)carve(wp, nX * 2);   // (B,H,DK,L)  transposed
  __bf16* ctx_b = (__bf16*)carve(wp, nX * 2);  // (B,L,D)
  float*  att_f = (float*)carve(wp, nX * 4);
  float*  x_f   = (float*)carve(wp, nX * 4);
  __bf16* x_b   = (__bf16*)carve(wp, nX * 2);
  __bf16* ffh_b = (__bf16*)carve(wp, nFF * 2);
  float*  ffo_f = (float*)carve(wp, nX * 4);

  const dim3 blk(256);
  auto cvt = [&](const float* src, __bf16* dst, size_t n) {
    cvt_f32_bf16<<<dim3((unsigned)((n + 255) / 256)), blk, 0, stream>>>(src, dst, (int)n);
  };
  auto cvtT = [&](const float* src, __bf16* dst, int K, int N) {
    cvt_transpose_bf16<<<dim3(N / 32, K / 32), dim3(32, 8), 0, stream>>>(src, dst, K, N);
  };
  cvt(query,  Xq_b, nX); cvt(key_in, Xk_b, nX); cvt(values, Xv_b, nX);
  cvtT(Wq, Wq_t, D_, D_); cvtT(Wk, Wk_t, D_, D_); cvtT(Wv, Wv_t, D_, D_);
  cvtT(Wo, Wo_t, D_, D_); cvtT(W1, W1_t, D_, DFF_); cvtT(W2, W2_t, DFF_, D_);

  const int M = B_ * L_;                       // 4096
  const dim3 gD(D_ / 256, M / 16);             // N=512  -> (2,256)
  const dim3 gF(DFF_ / 256, M / 16);           // N=2048 -> (8,256)

  // Q/K/V projections
  gemm_bf16_wmma<<<gD, blk, 0, stream>>>(Xq_b, Wq_t, bq, nullptr, q_b, M, D_, D_, 2);
  gemm_bf16_wmma<<<gD, blk, 0, stream>>>(Xk_b, Wk_t, bk, nullptr, k_b, M, D_, D_, 2);
  gemm_bf16_wmma<<<gD, blk, 0, stream>>>(Xv_b, Wv_t, bv, nullptr, v_t, M, D_, D_, 4);

  // fused forgetting-attention: 224 KB dynamic LDS (of 320 KB per WGP), K panel via TDM
  const size_t attn_lds = ATTN_KLDS_OFF + ATTN_K_BYTES;
  fused_attn<<<dim3(L_ / 16, B_ * H_), blk, attn_lds, stream>>>(q_b, k_b, v_t, satate, decay, ctx_b);

  // output projection -> f32
  gemm_bf16_wmma<<<gD, blk, 0, stream>>>(ctx_b, Wo_t, bo, att_f, nullptr, M, D_, D_, 0);

  // x = LN(query + att_out) -> f32 + bf16
  add_layernorm<<<dim3(M), blk, 0, stream>>>(query, att_f, ln1_g, ln1_b, x_f, x_b);

  // FFN: relu(x @ W1 + b1) @ W2 + b2
  gemm_bf16_wmma<<<gF, blk, 0, stream>>>(x_b, W1_t, b1, nullptr, ffh_b, M, DFF_, D_, 1);
  gemm_bf16_wmma<<<gD, blk, 0, stream>>>(ffh_b, W2_t, b2, ffo_f, nullptr, M, D_, DFF_, 0);

  // out = LN(x + ffn)
  add_layernorm<<<dim3(M), blk, 0, stream>>>(x_f, ffo_f, ln2_g, ln2_b, (float*)d_out, nullptr);
}
